// SpatialRBF_11879879540922
// MI455X (gfx1250) — matrix operand order repaired and checked by reading, not compiled, not run on hardware
//
#include <hip/hip_runtime.h>

typedef __attribute__((ext_vector_type(16))) _Float16 v16h;
typedef __attribute__((ext_vector_type(8)))  float    v8f;

#define NPTS   65536
#define GPTS   2304
#define OUTF   512
#define KCHUNK 128              // K staged per LDS round (4 WMMA k-steps of 32)
#define NCHUNK (GPTS / KCHUNK)  // 18
// exp(-80*d2) = exp2(-80*log2(e)*d2)
#define EXPCOEF (-115.41560327111707f)

// ---------------------------------------------------------------------------
// Pre-pass: spline_weight (G x OUT, f32, K-major) -> Bt (OUT x G, f16, N-major)
// so WMMA B fragments become contiguous 16B-aligned b128 loads.
// ---------------------------------------------------------------------------
__global__ __launch_bounds__(256) void convert_bt(const float* __restrict__ sw,
                                                  _Float16* __restrict__ Bt) {
  __shared__ _Float16 tile[32][33];
  const int k0 = blockIdx.x * 32;
  const int n0 = blockIdx.y * 32;
  const int tx = threadIdx.x & 31;
  const int ty = threadIdx.x >> 5;   // 0..7
#pragma unroll
  for (int i = 0; i < 32; i += 8) {
    tile[ty + i][tx] = (_Float16)sw[(size_t)(k0 + ty + i) * OUTF + (n0 + tx)];
  }
  __syncthreads();
#pragma unroll
  for (int i = 0; i < 32; i += 8) {
    Bt[(size_t)(n0 + ty + i) * GPTS + (k0 + tx)] = tile[tx][ty + i];
  }
}

// ---------------------------------------------------------------------------
// Fused RBF + GEMM:
//   out = x @ bwT  +  exp(-80*|x-grid|^2) @ spline_weight
// WG: 8 waves = 2 rowgroups(32 rows) x 4 colgroups(64 cols) -> 64x256 tile.
// Per 128-K chunk: waves cooperatively produce 16 A fragments (16x32 f16, in
// contiguous-k WMMA register layout) into double-buffered LDS, then each wave does
// 4 ksteps x 4 col-tiles x 2 row-tiles = 32 v_wmma_f32_16x16x32_f16.
// ---------------------------------------------------------------------------
__global__ __launch_bounds__(256) void rbf_wmma(
    const float* __restrict__ x, const float* __restrict__ grid,
    const _Float16* __restrict__ Bt, const float* __restrict__ bw,
    float* __restrict__ out) {
  // [buf][rowgroup 0..3][kstep 0..3][lane][2 x 16B] = 32 KB
  __shared__ uint4 ldsA[2][4][4][32][2];

  const int lane = threadIdx.x & 31;
  const int wave = threadIdx.x >> 5;
  const int half = lane >> 4;     // lane-half (K/M split selector)
  const int m    = lane & 15;

  const int rowBase = blockIdx.x * 64;
  const int colBase = blockIdx.y * 256;

  // consumer role: which 32 rows / 64 cols this wave owns
  const int rowHalf = wave >> 2;           // 0 or 1
  const int cg      = wave & 3;            // column group
  const int wcol    = colBase + cg * 64;

  // producer role: wave produces rowgroup prg, ksteps pks..pks+1
  const int prg  = wave >> 1;              // 0..3
  const int pks  = (wave & 1) * 2;         // 0 or 2
  const int prow = rowBase + prg * 16 + m; // A-fragment: lane holds row m
  const float px0 = x[2 * prow + 0];
  const float px1 = x[2 * prow + 1];

  v8f acc[2][4];
  // ---- init accumulators with the base linear term: x @ base_weight^T ----
#pragma unroll
  for (int rgi = 0; rgi < 2; ++rgi) {
    float x0[8], x1[8];
#pragma unroll
    for (int r = 0; r < 8; ++r) {
      // C/D layout: lanes 0-15 -> M=r, lanes 16-31 -> M=r+8
      int row = rowBase + (rowHalf * 2 + rgi) * 16 + (half ? r + 8 : r);
      x0[r] = x[2 * row + 0];
      x1[r] = x[2 * row + 1];
    }
#pragma unroll
    for (int t = 0; t < 4; ++t) {
      int n = wcol + t * 16 + m;
      float b0 = bw[2 * n + 0];
      float b1 = bw[2 * n + 1];
#pragma unroll
      for (int r = 0; r < 8; ++r) acc[rgi][t][r] = x0[r] * b0 + x1[r] * b1;
    }
  }

  for (int chunk = 0; chunk < NCHUNK; ++chunk) {
    const int buf = chunk & 1;
    const int kb  = chunk * KCHUNK;

    // ---- produce: 2 A fragments (16x32 f16) in contiguous-k WMMA layout ----
    // A 16x32 f16, contiguous k as used here (same as B): lane(0-15): V0..7 -> K=0..15 pairs;
    //                     lane(16-31): V0..7 -> K=16..31 pairs.
#pragma unroll
    for (int pf = 0; pf < 2; ++pf) {
      const int ks = pks + pf;
      union { _Float16 h[16]; uint4 q[2]; } af;
#pragma unroll
      for (int v = 0; v < 8; ++v) {
#pragma unroll
        for (int e = 0; e < 2; ++e) {
          int k = half * 16 + v * 2 + e;
          int g = kb + ks * 32 + k;
          float dx = px0 - grid[2 * g + 0];
          float dy = px1 - grid[2 * g + 1];
          af.h[2 * v + e] = (_Float16)exp2f(EXPCOEF * (dx * dx + dy * dy));
        }
      }
      ldsA[buf][prg][ks][lane][0] = af.q[0];
      ldsA[buf][prg][ks][lane][1] = af.q[1];
    }
    __syncthreads();  // waits DScnt; single barrier OK with double buffer

    // pull next chunk's B slab toward the caches
    if (chunk + 1 < NCHUNK)
      __builtin_prefetch(Bt + (size_t)(wcol + m) * GPTS + kb + KCHUNK, 0, 0);

    // ---- consume: 4 ksteps x 4 col tiles x 2 row tiles ----
#pragma unroll
    for (int ks = 0; ks < 4; ++ks) {
      union { v16h v; uint4 q[2]; } A0, A1;
      A0.q[0] = ldsA[buf][rowHalf * 2 + 0][ks][lane][0];
      A0.q[1] = ldsA[buf][rowHalf * 2 + 0][ks][lane][1];
      A1.q[0] = ldsA[buf][rowHalf * 2 + 1][ks][lane][0];
      A1.q[1] = ldsA[buf][rowHalf * 2 + 1][ks][lane][1];
      const int kc = kb + ks * 32;
#pragma unroll
      for (int t = 0; t < 4; ++t) {
        const int n = wcol + t * 16 + m;
        // B operand, same contiguous k as A: lanes 0-15 hold K=0..15, lanes 16-31 K=16..31,
        // pairs packed per VGPR -> 16 contiguous f16 per lane in Bt (N-major).
        const _Float16* bp = Bt + (size_t)n * GPTS + kc + 16 * half;
        union { v16h v; uint4 q[2]; } B;
        B.q[0] = *reinterpret_cast<const uint4*>(bp);
        B.q[1] = *reinterpret_cast<const uint4*>(bp + 8);
        acc[0][t] = __builtin_amdgcn_wmma_f32_16x16x32_f16(
            false, A0.v, false, B.v, (short)0, acc[0][t], false, false);
        acc[1][t] = __builtin_amdgcn_wmma_f32_16x16x32_f16(
            false, A1.v, false, B.v, (short)0, acc[1][t], false, false);
      }
    }
  }

  // ---- store D (f32), streaming (nontemporal) ----
#pragma unroll
  for (int rgi = 0; rgi < 2; ++rgi) {
#pragma unroll
    for (int t = 0; t < 4; ++t) {
      const int n = wcol + t * 16 + m;
#pragma unroll
      for (int r = 0; r < 8; ++r) {
        int row = rowBase + (rowHalf * 2 + rgi) * 16 + (half ? r + 8 : r);
        __builtin_nontemporal_store(acc[rgi][t][r], &out[(size_t)row * OUTF + n]);
      }
    }
  }
}

extern "C" void kernel_launch(void* const* d_in, const int* in_sizes, int n_in,
                              void* d_out, int out_size, void* d_ws, size_t ws_size,
                              hipStream_t stream) {
  const float* x    = (const float*)d_in[0];   // (N, 2)
  const float* grid = (const float*)d_in[1];   // (G, 2)
  const float* sw   = (const float*)d_in[2];   // (G, OUT)
  const float* bw   = (const float*)d_in[3];   // (OUT, 2)
  float* out = (float*)d_out;                  // (N, OUT)
  _Float16* Bt = (_Float16*)d_ws;              // (OUT, G) f16 transposed weights

  convert_bt<<<dim3(GPTS / 32, OUTF / 32), 256, 0, stream>>>(sw, Bt);
  rbf_wmma<<<dim3(NPTS / 64, OUTF / 256), 256, 0, stream>>>(x, grid, Bt, bw, out);
}
